// EfficientSatAttention_74998718923149
// MI455X (gfx1250) — matrix operand order repaired and checked by reading, not compiled, not run on hardware
//
#include <hip/hip_runtime.h>
#include <hip/hip_bf16.h>

// ---------------------------------------------------------------------------
// EfficientSatAttention for MI455X (gfx1250, wave32, WMMA)
//   K0: one-shot f32->f16 conversion of qkv_w / proj_w
//   K1: QKV GEMM  (x @ qkv_w^T + b) -> Q,K (token-major f16), V^T (dim-major)
//       k-loop fully unrolled so load clauses hoist across WMMAs
//   K2: fused flash attention, online softmax, per-(b,h) temperature,
//       confidence = mean(1/l), head-dropout keep scaling -> AO f16
//       K-fragments double-buffered (software pipeline, branch-free)
//   K3: projection GEMM + bias + learned residual blend   -> out f32
// All matmuls via v_wmma_f32_16x16x32_f16.
// ---------------------------------------------------------------------------

typedef __attribute__((ext_vector_type(16))) _Float16 v16h;
typedef __attribute__((ext_vector_type(8)))  _Float16 v8h;
typedef __attribute__((ext_vector_type(8)))  float    v8f;
typedef __attribute__((ext_vector_type(4)))  float    v4f;

#define WMMA(a, b, c) \
  __builtin_amdgcn_wmma_f32_16x16x32_f16(false, (a), false, (b), (short)0, (c), false, false)

static __device__ __forceinline__ v16h cat8(v8h a, v8h b) {
    return __builtin_shufflevector(a, b, 0, 1, 2, 3, 4, 5, 6, 7,
                                         8, 9, 10, 11, 12, 13, 14, 15);
}

// Problem constants
#define BB   32
#define NN   1024
#define CC   192
#define HH   3
#define DD   64
#define C3   576      // 3*C

// ---------------------------------------------------------------------------
// Kernel 0: weight conversion f32 -> f16 (qkv_w: 110592, proj_w: 36864 elems)
// ---------------------------------------------------------------------------
__global__ __launch_bounds__(256)
void sat_cvt_weights(const float* __restrict__ wq, const float* __restrict__ wp,
                     _Float16* __restrict__ wqh, _Float16* __restrict__ wph)
{
    const int i = (int)(blockIdx.x * blockDim.x + threadIdx.x);
    if (i < C3 * CC) wqh[i] = (_Float16)wq[i];
    if (i < CC * CC) wph[i] = (_Float16)wp[i];
}

// ---------------------------------------------------------------------------
// Kernel 1: QKV projection. One wave: 16(M) x 64(N) tile of (32768 x 576);
// K=192 fully unrolled (6 x 32). A-fragment (x, f32->f16) reused 4x.
// ---------------------------------------------------------------------------
__global__ __launch_bounds__(256)
void sat_qkv_gemm(const float* __restrict__ x, const _Float16* __restrict__ wh,
                  const float* __restrict__ bias,
                  _Float16* __restrict__ Qb, _Float16* __restrict__ Kb,
                  _Float16* __restrict__ Vt)
{
    const int wave = (int)((blockIdx.x * blockDim.x + threadIdx.x) >> 5);
    const int lane = (int)(threadIdx.x & 31);
    const int half = lane >> 4;
    const int ln   = lane & 15;

    const int NG = C3 / 64;            // 9 column groups of 64
    const int mt = wave / NG;          // 0..2047
    const int ng = wave % NG;          // 0..8
    const int m0 = mt * 16;
    const int n0 = ng * 64;
    const int row = m0 + ln;           // A-fragment: M = lane%16

    const float*    xbase = x  + (size_t)row * CC + half * 8;
    const _Float16* wbase = wh + (size_t)(n0 + ln) * CC + half * 8;

    v8f acc[4] = {v8f{}, v8f{}, v8f{}, v8f{}};

    #pragma unroll
    for (int k0 = 0; k0 < CC; k0 += 32) {
        const float* xr = xbase + k0;
        const v4f x0 = *(const v4f*)(xr +  0);
        const v4f x1 = *(const v4f*)(xr +  4);
        const v4f x2 = *(const v4f*)(xr + 16);
        const v4f x3 = *(const v4f*)(xr + 20);

        v8h blo[4], bhi[4];
        #pragma unroll
        for (int nt = 0; nt < 4; ++nt) {
            const _Float16* wr = wbase + (size_t)(nt * 16) * CC + k0;
            blo[nt] = *(const v8h*)(wr);
            bhi[nt] = *(const v8h*)(wr + 16);
        }

        v16h a;
        #pragma unroll
        for (int e = 0; e < 4; ++e) {
            a[e]      = (_Float16)x0[e];
            a[4 + e]  = (_Float16)x1[e];
            a[8 + e]  = (_Float16)x2[e];
            a[12 + e] = (_Float16)x3[e];
        }

        #pragma unroll
        for (int nt = 0; nt < 4; ++nt)
            acc[nt] = WMMA(a, cat8(blo[nt], bhi[nt]), acc[nt]);
    }

    #pragma unroll
    for (int nt = 0; nt < 4; ++nt) {
        const int col   = n0 + nt * 16 + ln;     // 0..575
        const int which = col / CC;              // 0=q 1=k 2=v
        const int cc    = col % CC;
        const int hh    = cc >> 6;
        const int dd    = cc & 63;
        const float bval = bias[col];
        #pragma unroll
        for (int v = 0; v < 8; ++v) {
            const int r  = m0 + v + 8 * half;    // D row m = v + 8*half
            const int bb = r >> 10;
            const int tn = r & (NN - 1);
            const _Float16 hv = (_Float16)(acc[nt][v] + bval);
            const size_t bh = (size_t)bb * HH + hh;
            if (which == 0)      Qb[(bh * NN + tn) * DD + dd] = hv;
            else if (which == 1) Kb[(bh * NN + tn) * DD + dd] = hv;
            else                 Vt[(bh * DD + dd) * NN + tn] = hv;
        }
    }
}

// ---------------------------------------------------------------------------
// Kernel 2: fused attention. One wave per (b, h, 16-query tile).
//   S^T = K_tile(16x64) @ Q^T(64x16): D-layout rows = keys, cols = queries.
//   Per-lane key sets of the S^T D tiles equal the A-fragment key sets for
//   P(16q x 32k) -> softmaxed P feeds PV WMMA with zero shuffles.
//   K fragments are double-buffered one iteration ahead (branch-free wrap).
// ---------------------------------------------------------------------------
struct KFrag { v8h a0, b0, a1, b1, a2, b2, a3, b3; };

static __device__ __forceinline__ KFrag load_kfrag(const _Float16* __restrict__ Krow,
                                                   int kt)
{
    const _Float16* kr0 = Krow + (size_t)kt * DD;
    const _Float16* kr1 = kr0 + 16 * DD;
    KFrag f;
    f.a0 = *(const v8h*)(kr0 +  0);  f.b0 = *(const v8h*)(kr0 + 16);
    f.a1 = *(const v8h*)(kr0 + 32);  f.b1 = *(const v8h*)(kr0 + 48);
    f.a2 = *(const v8h*)(kr1 +  0);  f.b2 = *(const v8h*)(kr1 + 16);
    f.a3 = *(const v8h*)(kr1 + 32);  f.b3 = *(const v8h*)(kr1 + 48);
    return f;
}

__global__ __launch_bounds__(256)
void sat_flash_attn(const _Float16* __restrict__ Qb, const _Float16* __restrict__ Kb,
                    const _Float16* __restrict__ Vt,
                    const float* __restrict__ temp_param,
                    const float* __restrict__ dropout_logits,
                    const float* __restrict__ tau_al,
                    _Float16* __restrict__ AO, float* __restrict__ conf)
{
    const int wave = (int)((blockIdx.x * blockDim.x + threadIdx.x) >> 5);
    const int lane = (int)(threadIdx.x & 31);
    const int half = lane >> 4;
    const int ln   = lane & 15;

    const int qt = wave & 63;          // 64 query tiles of 16
    const int bh = wave >> 6;          // 0..95
    const int b  = bh / HH;
    const int h  = bh % HH;
    const int q0 = qt * 16;

    // tau = tau_aleatoric[b,h] + (0.1 + softplus(p)) * anneal(=1.5)
    const float tp    = temp_param[h];
    const float tau_e = (0.1f + logf(1.0f + __expf(tp))) * 1.5f;
    const float tau   = tau_al[b * HH + h] + tau_e;
    const float coef  = 0.125f / tau;                      // D^-0.5 / tau
    const float keep  = 1.0f - 0.3f / (1.0f + __expf(-dropout_logits[h]));

    // Q B-fragments (K = head dim): lane n = query, k = c*32 + half*8 + e (e<8), c*32 + 16 + half*8 + (e-8) (e>=8)
    const _Float16* Qrow = Qb + ((size_t)bh * NN + q0 + ln) * DD + half * 8;
    const v16h qb0 = cat8(*(const v8h*)(Qrow),      *(const v8h*)(Qrow + 16));
    const v16h qb1 = cat8(*(const v8h*)(Qrow + 32), *(const v8h*)(Qrow + 48));

    const _Float16* Krow  = Kb + ((size_t)bh * NN + ln) * DD + half * 8;
    const _Float16* Vbase = Vt + (size_t)bh * DD * NN + (size_t)ln * NN + half * 8;

    float m_i = -1e30f, l_i = 0.0f;
    v8f o[4] = {v8f{}, v8f{}, v8f{}, v8f{}};

    // software pipeline: K fragments for tile 0 in flight before the loop
    KFrag kc = load_kfrag(Krow, 0);

    for (int kt = 0; kt < NN; kt += 32) {
        // next K tile (wraps to 0 on the last iteration -> branch-free, L2 hit)
        KFrag kn = load_kfrag(Krow, (kt + 32) & (NN - 1));

        // V fragments for this tile: consumed only after S WMMAs + softmax
        v8h vblo[4], vbhi[4];
        #pragma unroll
        for (int nt = 0; nt < 4; ++nt) {
            const _Float16* vr = Vbase + (size_t)(nt * 16) * NN + kt;
            vblo[nt] = *(const v8h*)(vr);
            vbhi[nt] = *(const v8h*)(vr + 16);
        }

        // --- S^T = K @ Q^T (current-buffer K fragments) ---
        v8f s0 = v8f{}; s0 = WMMA(cat8(kc.a0, kc.b0), qb0, s0);
                        s0 = WMMA(cat8(kc.a1, kc.b1), qb1, s0);
        v8f s1 = v8f{}; s1 = WMMA(cat8(kc.a2, kc.b2), qb0, s1);
                        s1 = WMMA(cat8(kc.a3, kc.b3), qb1, s1);

        // scale + per-query max: lane holds 16 of 32 keys, partner the rest
        float tmax = -1e30f;
        #pragma unroll
        for (int v = 0; v < 8; ++v) {
            s0[v] *= coef; s1[v] *= coef;
            tmax = fmaxf(tmax, fmaxf(s0[v], s1[v]));
        }
        tmax = fmaxf(tmax, __shfl_xor(tmax, 16, 32));
        const float m_new = fmaxf(m_i, tmax);
        const float alpha = __expf(m_i - m_new);

        float psum = 0.0f;
        #pragma unroll
        for (int v = 0; v < 8; ++v) {
            s0[v] = __expf(s0[v] - m_new);
            s1[v] = __expf(s1[v] - m_new);
            psum += s0[v] + s1[v];
        }
        psum += __shfl_xor(psum, 16, 32);
        l_i = l_i * alpha + psum;
        m_i = m_new;

        #pragma unroll
        for (int nt = 0; nt < 4; ++nt)
            #pragma unroll
            for (int v = 0; v < 8; ++v) o[nt][v] *= alpha;

        // P A-fragment: pure register remap from S^T D tiles
        v16h pa;
        #pragma unroll
        for (int v = 0; v < 8; ++v) { pa[v] = (_Float16)s0[v]; pa[8 + v] = (_Float16)s1[v]; }

        #pragma unroll
        for (int nt = 0; nt < 4; ++nt)
            o[nt] = WMMA(pa, cat8(vblo[nt], vbhi[nt]), o[nt]);

        kc = kn;   // rotate pipeline (pure register renaming)
    }

    // Finalize: O row m = query (v + 8*half); softmax stats live at lane == query
    #pragma unroll
    for (int v = 0; v < 8; ++v) {
        const int qrow  = v + 8 * half;
        const float lrw = __shfl(l_i, qrow, 32);
        const float sc  = keep / lrw;
        _Float16* aor = AO + ((size_t)b * NN + q0 + qrow) * CC + h * DD;
        #pragma unroll
        for (int nt = 0; nt < 4; ++nt)
            aor[nt * 16 + ln] = (_Float16)(o[nt][v] * sc);
    }

    // confidence: max prob per row = 1/l; lanes l and l^16 hold duplicates
    float cinv = 1.0f / l_i;
    cinv += __shfl_xor(cinv, 1, 32);
    cinv += __shfl_xor(cinv, 2, 32);
    cinv += __shfl_xor(cinv, 4, 32);
    cinv += __shfl_xor(cinv, 8, 32);
    cinv += __shfl_xor(cinv, 16, 32);
    if (lane == 0) atomicAdd(&conf[b], cinv * (0.5f / (float)(HH * NN)));
}

// ---------------------------------------------------------------------------
// Kernel 3: output projection + bias + learned residual blend.
// One wave: 16(M) x 64(N) tile of (32768 x 192); K fully unrolled; f16 inputs.
// ---------------------------------------------------------------------------
__global__ __launch_bounds__(256)
void sat_proj_blend(const _Float16* __restrict__ AO, const _Float16* __restrict__ wh,
                    const float* __restrict__ bias, const float* __restrict__ x,
                    const float* __restrict__ resid_p, float* __restrict__ out)
{
    const int wave = (int)((blockIdx.x * blockDim.x + threadIdx.x) >> 5);
    const int lane = (int)(threadIdx.x & 31);
    const int half = lane >> 4;
    const int ln   = lane & 15;

    const int NG = CC / 64;            // 3 column groups
    const int mt = wave / NG;
    const int ng = wave % NG;
    const int m0 = mt * 16;
    const int n0 = ng * 64;
    const int row = m0 + ln;

    const float wres = 1.0f / (1.0f + __expf(-resid_p[0]));

    const _Float16* abase = AO + (size_t)row * CC + half * 8;
    const _Float16* wbase = wh + (size_t)(n0 + ln) * CC + half * 8;

    v8f acc[4] = {v8f{}, v8f{}, v8f{}, v8f{}};

    #pragma unroll
    for (int k0 = 0; k0 < CC; k0 += 32) {
        const v8h alo = *(const v8h*)(abase + k0);
        const v8h ahi = *(const v8h*)(abase + k0 + 16);

        v8h blo[4], bhi[4];
        #pragma unroll
        for (int nt = 0; nt < 4; ++nt) {
            const _Float16* wr = wbase + (size_t)(nt * 16) * CC + k0;
            blo[nt] = *(const v8h*)(wr);
            bhi[nt] = *(const v8h*)(wr + 16);
        }

        const v16h a = cat8(alo, ahi);
        #pragma unroll
        for (int nt = 0; nt < 4; ++nt)
            acc[nt] = WMMA(a, cat8(blo[nt], bhi[nt]), acc[nt]);
    }

    #pragma unroll
    for (int nt = 0; nt < 4; ++nt) {
        const int col = n0 + nt * 16 + ln;
        const float bval = bias[col];
        #pragma unroll
        for (int v = 0; v < 8; ++v) {
            const int r = m0 + v + 8 * half;
            const size_t idx = (size_t)r * CC + col;
            out[idx] = wres * (acc[nt][v] + bval) + (1.0f - wres) * x[idx];
        }
    }
}

// ---------------------------------------------------------------------------
extern "C" void kernel_launch(void* const* d_in, const int* in_sizes, int n_in,
                              void* d_out, int out_size, void* d_ws, size_t ws_size,
                              hipStream_t stream)
{
    (void)in_sizes; (void)n_in; (void)out_size; (void)ws_size;

    const float* x              = (const float*)d_in[0];
    const float* qkv_w          = (const float*)d_in[1];
    const float* qkv_b          = (const float*)d_in[2];
    const float* proj_w         = (const float*)d_in[3];
    const float* proj_b         = (const float*)d_in[4];
    const float* temp_param     = (const float*)d_in[5];
    const float* dropout_logits = (const float*)d_in[6];
    const float* residual_param = (const float*)d_in[7];
    const float* tau_aleatoric  = (const float*)d_in[8];

    float* out  = (float*)d_out;
    float* conf = out + (size_t)BB * NN * CC;   // confidence tail (32 floats)

    // Workspace: Q, K (token-major f16), V^T (dim-major f16), AO (f16),
    // then f16 weight copies.
    const size_t elems = (size_t)BB * HH * NN * DD;   // 6,291,456
    _Float16* Qb  = (_Float16*)d_ws;
    _Float16* Kb  = Qb + elems;
    _Float16* Vt  = Kb + elems;
    _Float16* AO  = Vt + elems;
    _Float16* wqh = AO + elems;
    _Float16* wph = wqh + (size_t)C3 * CC;

    hipMemsetAsync(conf, 0, BB * sizeof(float), stream);

    // K0: weight conversion (110592 elems -> 432 blocks of 256)
    sat_cvt_weights<<<432, 256, 0, stream>>>(qkv_w, proj_w, wqh, wph);

    // K1: 2048 M-tiles * 9 N-groups = 18432 waves -> 2304 blocks of 8 waves
    sat_qkv_gemm<<<2304, 256, 0, stream>>>(x, wqh, qkv_b, Qb, Kb, Vt);

    // K2: 32*3 heads * 64 query tiles = 6144 waves -> 768 blocks
    sat_flash_attn<<<768, 256, 0, stream>>>(Qb, Kb, Vt, temp_param, dropout_logits,
                                            tau_aleatoric, AO, conf);

    // K3: 2048 M-tiles * 3 N-groups = 6144 waves -> 768 blocks
    sat_proj_blend<<<768, 256, 0, stream>>>(AO, wph, proj_b, x, residual_param, out);
}
